// QueryPromptAugmented_65996467470350
// MI455X (gfx1250) — matrix operand order repaired and checked by reading, not compiled, not run on hardware
//
#include <hip/hip_runtime.h>
#include <hip/hip_bf16.h>

// ---------------------------------------------------------------------------
// QueryPromptAugmented forward for MI455X (gfx1250, wave32).
// Memory-bound (~1 GB traffic -> ~41 us at 23.3 TB/s); the tiny 80x80x128
// t@t.T is mapped onto v_wmma_f32_16x16x32_f16 (f16 in, f32 accumulate).
// Big streams use 128-bit non-temporal loads/stores (1 GB >> 192 MB L2).
// ---------------------------------------------------------------------------

typedef __attribute__((ext_vector_type(16))) _Float16 v16h;
typedef __attribute__((ext_vector_type(8)))  float    v8f;
typedef __attribute__((ext_vector_type(4)))  float    f4;
typedef __attribute__((ext_vector_type(4)))  int      i4;

#define TOKDIM 128
#define TMAX   80          // A + 2V = 64 + 16

// ---------------------------------------------------------------------------
// Kernel 1: one workgroup (256 threads = 8 wave32s).
//   - build t = [token_list[attr_index]; zeros(V); vir_list] in LDS (f16) and
//     store the f32 rows into x_cat[0:T]
//   - token_sim = sigmoid(t @ t.T) via WMMA 16x16x32 f16, inner_adj threshold
//   - copy query
// ---------------------------------------------------------------------------
__global__ void __launch_bounds__(256)
token_sim_kernel(const float* __restrict__ token_list,
                 const float* __restrict__ vir_list,
                 const int*   __restrict__ attr_index,
                 const int*   __restrict__ query,
                 float* __restrict__ xcat,   // d_out head: rows 0..T-1
                 float* __restrict__ sim,    // token_sim [T,T]
                 float* __restrict__ adj,    // inner_adj [T,T]
                 float* __restrict__ qout,   // query copy
                 int A, int V, int T, int NQ)
{
    __shared__ _Float16 t16[TMAX * TOKDIM];   // 20 KB of the 320 KB WGP LDS

    const int tid = threadIdx.x;

    // Build t (f32 -> x_cat head, f16 -> LDS). T*TOKDIM = 10240 = 40*256,
    // so the loop trip count is wave-uniform (no EXEC divergence).
    for (int idx = tid; idx < T * TOKDIM; idx += (int)blockDim.x) {
        const int r = idx / TOKDIM;
        const int k = idx - r * TOKDIM;
        float v;
        if (r < A)          v = token_list[(size_t)attr_index[r] * TOKDIM + k];
        else if (r < A + V) v = 0.0f;                     // zeros block
        else                v = vir_list[(size_t)(r - A - V) * TOKDIM + k];
        t16[idx]  = (_Float16)v;
        xcat[idx] = v;
    }
    __syncthreads();

    const int wave  = tid >> 5;
    const int lane  = tid & 31;
    const int hsel  = lane >> 4;    // 0: lanes 0-15, 1: lanes 16-31
    const int l16   = lane & 15;
    const int nt    = T >> 4;       // 5 tiles per side
    const int nwave = (int)blockDim.x >> 5;

    // Each wave computes whole 16x16 output tiles (EXEC all-ones at WMMA).
    for (int tile = wave; tile < nt * nt; tile += nwave) {
        const int mi = tile / nt, ni = tile - mi * nt;
        const int mr = mi << 4,  nr = ni << 4;

        // A-operand: row M = mr + (lane&15); K halves per ISA 16-bit A layout.
        const int rowA  = mr + l16;
        const int koffA = hsel * 8;     // lanes 16-31 start at K=8 (then K=24)
        // B-operand: B[k][n] = t[n][k]; lane holds column N = nr + (lane&15).
        const int rowB  = nr + l16;
        const int koffB = hsel * 8;     // lanes 16-31 start at K=8 (then K=24)

        v8f c = {};
        #pragma unroll
        for (int kc = 0; kc < TOKDIM; kc += 32) {
            v16h a, b;
            #pragma unroll
            for (int j = 0; j < 8; ++j) {
                a[j]     = t16[rowA * TOKDIM + kc + koffA + j];
                a[j + 8] = t16[rowA * TOKDIM + kc + koffA + 16 + j];
            }
            #pragma unroll
            for (int j = 0; j < 8; ++j) {
                b[j]     = t16[rowB * TOKDIM + kc + koffB + j];
                b[j + 8] = t16[rowB * TOKDIM + kc + koffB + 16 + j];
            }
            // D = A x B + C  (f32 accumulate)
            c = __builtin_amdgcn_wmma_f32_16x16x32_f16(
                    /*neg_a=*/false, a, /*neg_b=*/false, b,
                    /*c_mod=*/(short)0, c, /*reuse_a=*/false, /*reuse_b=*/false);
        }

        // C/D layout: VGPR r -> row mr + r + 8*hsel, col nr + (lane&15)
        #pragma unroll
        for (int r = 0; r < 8; ++r) {
            const int gm = mr + r + hsel * 8;
            const int gn = nr + l16;
            const float s = 1.0f / (1.0f + __expf(-c[r]));   // sigmoid
            sim[gm * T + gn] = s;
            adj[gm * T + gn] = (s >= 0.01f) ? 1.0f : 0.0f;   // INNER_PRUNE
        }
    }

    if (tid < NQ) qout[tid] = (float)query[tid];
}

// ---------------------------------------------------------------------------
// Kernel 2: stream x (N*128 f32) into x_cat[T:]. 128-bit NT load/store.
// ---------------------------------------------------------------------------
__global__ void __launch_bounds__(256)
copy_x_kernel(const float* __restrict__ x, float* __restrict__ dst, long long n4)
{
    const long long i = (long long)blockIdx.x * blockDim.x + threadIdx.x;
    if (i < n4) {
        f4 v = __builtin_nontemporal_load((const f4*)x + i);
        __builtin_nontemporal_store(v, (f4*)dst + i);
    }
}

// ---------------------------------------------------------------------------
// Kernel 3: shifted_edge_index = edge_index + T (int -> float output stream).
// ---------------------------------------------------------------------------
__global__ void __launch_bounds__(256)
edge_shift_kernel(const int* __restrict__ e, float* __restrict__ o,
                  long long n, long long n4, int T)
{
    const long long i = (long long)blockIdx.x * blockDim.x + threadIdx.x;
    if (i < n4) {
        i4 v = __builtin_nontemporal_load((const i4*)e + i);
        f4 r;
        r[0] = (float)(v[0] + T);
        r[1] = (float)(v[1] + T);
        r[2] = (float)(v[2] + T);
        r[3] = (float)(v[3] + T);
        __builtin_nontemporal_store(r, (f4*)o + i);
    }
    const long long base = n4 * 4;          // scalar tail (normally empty)
    if (i < n - base) o[base + i] = (float)(e[base + i] + T);
}

// ---------------------------------------------------------------------------
// Kernel 4: attr_mask = (x_attr != 0) as 0.0/1.0 float stream.
// ---------------------------------------------------------------------------
__global__ void __launch_bounds__(256)
attr_mask_kernel(const int* __restrict__ a, float* __restrict__ o,
                 long long n, long long n4)
{
    const long long i = (long long)blockIdx.x * blockDim.x + threadIdx.x;
    if (i < n4) {
        i4 v = __builtin_nontemporal_load((const i4*)a + i);
        f4 r;
        r[0] = (v[0] != 0) ? 1.0f : 0.0f;
        r[1] = (v[1] != 0) ? 1.0f : 0.0f;
        r[2] = (v[2] != 0) ? 1.0f : 0.0f;
        r[3] = (v[3] != 0) ? 1.0f : 0.0f;
        __builtin_nontemporal_store(r, (f4*)o + i);
    }
    const long long base = n4 * 4;          // scalar tail (normally empty)
    if (i < n - base) o[base + i] = (a[base + i] != 0) ? 1.0f : 0.0f;
}

// ---------------------------------------------------------------------------
extern "C" void kernel_launch(void* const* d_in, const int* in_sizes, int n_in,
                              void* d_out, int out_size, void* d_ws, size_t ws_size,
                              hipStream_t stream)
{
    // Inputs (setup_inputs order): x, token_list, vir_list, attr_index,
    //                              x_attr, query, edge_index
    const float* x        = (const float*)d_in[0];
    const float* token_l  = (const float*)d_in[1];
    const float* vir_l    = (const float*)d_in[2];
    const int*   attr_idx = (const int*)  d_in[3];
    const int*   x_attr   = (const int*)  d_in[4];
    const int*   query    = (const int*)  d_in[5];
    const int*   edge     = (const int*)  d_in[6];

    const long long Nd = in_sizes[0];            // N * 128
    const int A  = in_sizes[3];                  // 64
    const int V  = in_sizes[2] / TOKDIM;         // 8
    const int T  = A + 2 * V;                    // 80
    const long long NA = in_sizes[4];            // N * A
    const int NQ = in_sizes[5];                  // 32
    const long long E2 = in_sizes[6];            // 2 * E

    // Output layout (flat, return order):
    //   x_cat[(T+N)*128] | token_sim[T*T] | inner_adj[T*T] |
    //   shifted_edge[2E] | attr_mask[N*A] | query[NQ]
    float* out   = (float*)d_out;
    float* xcat  = out;
    float* sim   = xcat + ((long long)T * TOKDIM + Nd);
    float* adj   = sim  + (long long)T * T;
    float* eout  = adj  + (long long)T * T;
    float* amask = eout + E2;
    float* qout  = amask + NA;

    // 1) tiny token/WMMA kernel: one workgroup
    token_sim_kernel<<<1, 256, 0, stream>>>(token_l, vir_l, attr_idx, query,
                                            xcat, sim, adj, qout, A, V, T, NQ);

    // 2) x -> x_cat tail (64M floats, vec4 NT stream)
    {
        const long long n4 = Nd / 4;
        const int blocks = (int)((n4 + 255) / 256);
        copy_x_kernel<<<blocks, 256, 0, stream>>>(
            x, xcat + (long long)T * TOKDIM, n4);
    }

    // 3) edge_index + T (16M elements)
    {
        const long long n4 = E2 / 4;
        const int blocks = (int)((n4 + 255) / 256);
        edge_shift_kernel<<<blocks, 256, 0, stream>>>(edge, eout, E2, n4, T);
    }

    // 4) attr_mask (32M elements)
    {
        const long long n4 = NA / 4;
        const int blocks = (int)((n4 + 255) / 256);
        attr_mask_kernel<<<blocks, 256, 0, stream>>>(x_attr, amask, NA, n4);
    }
}
